// Attention_5334349381821
// MI455X (gfx1250) — matrix operand order repaired and checked by reading, not compiled, not run on hardware
//
#include <hip/hip_runtime.h>
#include <hip/hip_bf16.h>

// Problem constants (match reference setup_inputs)
#define DIMX     1024
#define SEQL     2048
#define BATCH    2
#define NHEADS   16
#define HD       64
#define NROWS    (BATCH * SEQL)          // 4096 token rows
#define RMS_EPS  1.1920929e-07f

typedef __attribute__((ext_vector_type(16))) _Float16 v16h;
typedef __attribute__((ext_vector_type(8)))  float    v8f;

// CDNA5 async global->LDS path (guarded: falls back to plain LDS staging)
#if defined(__AMDGCN__) && __has_builtin(__builtin_amdgcn_global_load_async_to_lds_b128) && \
    __has_builtin(__builtin_amdgcn_s_wait_asynccnt)
#define USE_ASYNC_LDS 1
typedef int async_i4 __attribute__((vector_size(4 * sizeof(int))));
#define AS1P(x) ((__attribute__((address_space(1))) async_i4*)(x))
#define AS3P(x) ((__attribute__((address_space(3))) async_i4*)(x))
#else
#define USE_ASYNC_LDS 0
#endif

union Frag {
  v16h     h;
  unsigned u[8];
};

__device__ __forceinline__ v8f wmma_f16(const Frag& a, const Frag& b, v8f c) {
  // D = A(16x32 f16) * B(32x16 f16) + C(16x16 f32)
  return __builtin_amdgcn_wmma_f32_16x16x32_f16(
      /*neg_a=*/false, a.h, /*neg_b=*/false, b.h,
      /*c_mod=*/(short)0, c, /*reuse_a=*/false, /*reuse_b=*/false);
}

// ---- A-fragment (16x32 f16) from row-major f16 matrix [rows, ld] ----
// ISA 7.12.2: lane m=lane&15 (both halves hold M=0..15); h=lane>>4.
// VGPR v: K = 16*h + 2*v, +1 (contiguous order chosen here, not the ISA table order; B uses the same)
__device__ __forceinline__ Frag load_a(const _Float16* p, int ld, int row0,
                                       int kbase, int lane) {
  Frag f;
  const int m = lane & 15, h = lane >> 4;
  const _Float16* r = p + (size_t)(row0 + m) * ld + kbase + 16 * h;
#pragma unroll
  for (int v = 0; v < 8; ++v) f.u[v] = *(const unsigned*)(r + 2 * v);
  return f;
}

// ---- B-fragment (32x16 f16), B(k,n) = W[ncol0+n][kbase+k] from row-major W ----
// lanes 0-15 hold K=0..15, lanes 16-31 hold K=16..31; VGPR v: K = 16*h + 2*v, +1
__device__ __forceinline__ Frag load_bt(const _Float16* p, int ld, int ncol0,
                                        int kbase, int lane) {
  Frag f;
  const int n = lane & 15, h = lane >> 4;
  const _Float16* r = p + (size_t)(ncol0 + n) * ld + kbase + 16 * h;
#pragma unroll
  for (int v = 0; v < 8; ++v) f.u[v] = *(const unsigned*)(r + 2 * v);
  return f;
}

// ============================ Kernel 0: weights fp32 -> f16 ==================
__global__ void cvt_weights(const float* __restrict__ wq,
                            const float* __restrict__ wk,
                            const float* __restrict__ wv,
                            const float* __restrict__ wo,
                            _Float16* __restrict__ dst) {
  const int i = blockIdx.x * 256 + threadIdx.x;   // 4 * 2^20 threads
  const int which = i >> 20;
  const int off = i & ((1 << 20) - 1);
  float v;
  if (which == 0)      v = wq[off];
  else if (which == 1) v = wk[off];
  else if (which == 2) v = wv[off];
  else                 v = wo[off];
  dst[i] = (_Float16)v;
}

// ============================ Kernel 1: RMSNorm -> f16 =======================
__global__ void rmsnorm(const float* __restrict__ x, const float* __restrict__ w,
                        _Float16* __restrict__ xh) {
  __shared__ float red[256];
  const int row = blockIdx.x;
  const float* xr = x + (size_t)row * DIMX;
  float s = 0.f;
  for (int i = threadIdx.x; i < DIMX; i += 256) { float v = xr[i]; s += v * v; }
  red[threadIdx.x] = s;
  __syncthreads();
  for (int st = 128; st > 0; st >>= 1) {
    if ((int)threadIdx.x < st) red[threadIdx.x] += red[threadIdx.x + st];
    __syncthreads();
  }
  const float inv = rsqrtf(red[0] * (1.0f / DIMX) + RMS_EPS);
  for (int i = threadIdx.x; i < DIMX; i += 256)
    xh[(size_t)row * DIMX + i] = (_Float16)(xr[i] * inv * w[i]);
}

// ============== Kernel 2: QKV projection GEMM (Y = xh @ W^T) =================
// One wave computes a 32x64 tile (two 16-row sub-tiles share each B fragment).
// Writes f16 in [b, h, n, d] layout; q gets the 1/sqrt(64) scale.
__global__ void qkv_gemm(const _Float16* __restrict__ xh,
                         const _Float16* __restrict__ wh,   // wq|wk|wv f16 contiguous
                         _Float16* __restrict__ qh,
                         _Float16* __restrict__ kh,
                         _Float16* __restrict__ vh) {
  const int wave = blockIdx.x * (blockDim.x >> 5) + (threadIdx.x >> 5);
  const int lane = threadIdx.x & 31;
  const int RT = NROWS / 32;           // 128 row tiles (32 rows each)
  const int CG = DIMX / HD;            // 16 col groups (== heads)
  const int mat = wave / (RT * CG);
  const int rem = wave % (RT * CG);
  const int rt = rem / CG;
  const int cg = rem % CG;
  const _Float16* W = wh + (size_t)mat * DIMX * DIMX;

  v8f c[2][4] = {};
  for (int kb = 0; kb < DIMX; kb += 32) {
    // prefetch next weight / activation cachelines into the WGP while WMMAs run
    __builtin_prefetch(W + (size_t)(cg * HD + (lane & 15)) * DIMX + kb + 128, 0, 0);
    __builtin_prefetch(xh + (size_t)(rt * 32 + (lane & 31)) * DIMX + kb + 128, 0, 0);
    Frag a0 = load_a(xh, DIMX, rt * 32, kb, lane);
    Frag a1 = load_a(xh, DIMX, rt * 32 + 16, kb, lane);
#pragma unroll
    for (int nt = 0; nt < 4; ++nt) {
      Frag b = load_bt(W, DIMX, cg * HD + nt * 16, kb, lane);
      c[0][nt] = wmma_f16(a0, b, c[0][nt]);
      c[1][nt] = wmma_f16(a1, b, c[1][nt]);
    }
  }

  _Float16* dst = (mat == 0) ? qh : (mat == 1) ? kh : vh;
  const float scale = (mat == 0) ? 0.125f : 1.0f;  // DIM_HEAD^-0.5 (rotation-commuting)
  const int h = lane >> 4, col = lane & 15;
#pragma unroll
  for (int st = 0; st < 2; ++st) {
#pragma unroll
    for (int nt = 0; nt < 4; ++nt) {
#pragma unroll
      for (int j = 0; j < 8; ++j) {
        const int r = rt * 32 + st * 16 + j + 8 * h;   // global token row
        const int bb = r / SEQL, n = r % SEQL;
        const int d = nt * 16 + col;
        dst[(((size_t)(bb * NHEADS + cg) * SEQL) + n) * HD + d] =
            (_Float16)(c[st][nt][j] * scale);
      }
    }
  }
}

// ============================ Kernel 3: RoPE on q/k ==========================
__global__ void rope_kernel(_Float16* __restrict__ qh, _Float16* __restrict__ kh) {
  const int t = blockIdx.x * 256 + threadIdx.x;   // over B*H*SEQ*32 pairs
  const int pair = t & 31;
  const int row = t >> 5;                         // [0, B*H*SEQ)
  const int n = row % SEQL;
  _Float16* base = ((blockIdx.y == 0) ? qh : kh) + (size_t)row * HD + pair * 2;
  const float inv = powf(10000.0f, -(float)pair / 32.0f);  // theta^-(2i/64)
  const float f = (float)n * inv;
  const float cs = cosf(f), sn = sinf(f);
  const float x0 = (float)base[0], x1 = (float)base[1];
  base[0] = (_Float16)(x0 * cs - x1 * sn);
  base[1] = (_Float16)(x1 * cs + x0 * sn);
}

// ==================== Kernel 4: causal flash attention =======================
// One wave per (b*h, 16-query tile); 32 keys per iteration so the PV step uses
// full K=32 WMMAs. V tile (4 KB, contiguous) staged to LDS — via CDNA5 async
// global->LDS when available (overlapped with the QK^T WMMAs + softmax).
__global__ void attn_kernel(const _Float16* __restrict__ qh,
                            const _Float16* __restrict__ kh,
                            const _Float16* __restrict__ vh,
                            _Float16* __restrict__ ao) {
  __shared__ _Float16 vtile[32 * HD];   // [key][d], 4 KB
  __shared__ _Float16 pLds[16 * 32];    // P tile, 1 KB
  const int lane = threadIdx.x & 31;
  const int qt = blockIdx.x;            // query tile
  const int bh = blockIdx.y;            // b*NHEADS + h
  const _Float16* Q = qh + (size_t)bh * SEQL * HD;
  const _Float16* K = kh + (size_t)bh * SEQL * HD;
  const _Float16* V = vh + (size_t)bh * SEQL * HD;
  const int h = lane >> 4;

  const Frag aq0 = load_a(Q, HD, qt * 16, 0, lane);
  const Frag aq1 = load_a(Q, HD, qt * 16, 32, lane);

  v8f o[4] = {};
  float mrow[8], l[8];
#pragma unroll
  for (int j = 0; j < 8; ++j) { mrow[j] = -3.4028235e38f; l[j] = 0.f; }

  const int nkb = (qt + 2) >> 1;        // ceil((qt+1)*16 / 32) key blocks
  for (int kb = 0; kb < nkb; ++kb) {
    const int key0 = kb * 32;

    // ---- stage V[key0..key0+31][0..63] into LDS (contiguous 4 KB) ----
#if USE_ASYNC_LDS
    {
      const char* gs = (const char*)(V + (size_t)key0 * HD);
      char* ls = (char*)vtile;
#pragma unroll
      for (int t = 0; t < 8; ++t) {
        const int ofs = (t * 32 + lane) * 16;
        __builtin_amdgcn_global_load_async_to_lds_b128(AS1P(gs + ofs), AS3P(ls + ofs), 0, 0);
      }
    }
#else
    {
      const uint4* gs = (const uint4*)(V + (size_t)key0 * HD);
      uint4* ls = (uint4*)vtile;
      for (int i = lane; i < 256; i += 32) ls[i] = gs[i];
    }
#endif

    // ---- S = Q(16x64) * K^T for two 16-key sub-tiles ----
    Frag bkA0 = load_bt(K, HD, key0, 0, lane);
    Frag bkA1 = load_bt(K, HD, key0, 32, lane);
    Frag bkB0 = load_bt(K, HD, key0 + 16, 0, lane);
    Frag bkB1 = load_bt(K, HD, key0 + 16, 32, lane);
    v8f s0 = {}, s1 = {};
    s0 = wmma_f16(aq0, bkA0, s0);
    s0 = wmma_f16(aq1, bkA1, s0);
    s1 = wmma_f16(aq0, bkB0, s1);
    s1 = wmma_f16(aq1, bkB1, s1);

    // ---- causal mask + online softmax over 32 keys ----
    const int kj0 = key0 + (lane & 15);
    const int kj1 = kj0 + 16;
#pragma unroll
    for (int jv = 0; jv < 8; ++jv) {
      const int qi = qt * 16 + jv + 8 * h;
      float sv0 = (kj0 <= qi) ? s0[jv] : -3.4028235e38f;
      float sv1 = (kj1 <= qi) ? s1[jv] : -3.4028235e38f;
      float rm = fmaxf(sv0, sv1);       // row reduce within 16-lane half
      rm = fmaxf(rm, __shfl_xor(rm, 1, 32));
      rm = fmaxf(rm, __shfl_xor(rm, 2, 32));
      rm = fmaxf(rm, __shfl_xor(rm, 4, 32));
      rm = fmaxf(rm, __shfl_xor(rm, 8, 32));
      const float newm = fmaxf(mrow[jv], rm);
      const float alpha = expf(mrow[jv] - newm);
      const float p0 = expf(sv0 - newm);
      const float p1 = expf(sv1 - newm);
      float rs = p0 + p1;
      rs += __shfl_xor(rs, 1, 32);
      rs += __shfl_xor(rs, 2, 32);
      rs += __shfl_xor(rs, 4, 32);
      rs += __shfl_xor(rs, 8, 32);
      l[jv] = l[jv] * alpha + rs;
      mrow[jv] = newm;
#pragma unroll
      for (int nt = 0; nt < 4; ++nt) o[nt][jv] *= alpha;
      const int m = jv + 8 * h;
      pLds[m * 32 + (lane & 15)] = (_Float16)p0;
      pLds[m * 32 + 16 + (lane & 15)] = (_Float16)p1;
    }

#if USE_ASYNC_LDS
    __builtin_amdgcn_s_wait_asynccnt(0);
#endif
    __syncthreads();                    // vtile + pLds visible to all lanes

    // ---- O += P(16x32) * V(32x64), full K=32 WMMAs ----
    const Frag pa = load_a(pLds, 32, 0, 0, lane);
    const unsigned short* vt = (const unsigned short*)vtile;
    const int dlo = lane & 15;
#pragma unroll
    for (int nt = 0; nt < 4; ++nt) {
      Frag bv;
      const int dcol = nt * 16 + dlo;
#pragma unroll
      for (int v = 0; v < 8; ++v) {
        const int k0 = 16 * h + 2 * v;
        bv.u[v] = (unsigned)vt[k0 * HD + dcol] |
                  ((unsigned)vt[(k0 + 1) * HD + dcol] << 16);
      }
      o[nt] = wmma_f16(pa, bv, o[nt]);
    }
    __syncthreads();                    // safe to restage vtile/pLds next iter
  }

  // epilogue: normalize and store f16 into [b, n, h*64+d]
  const int bb = bh / NHEADS, hh = bh % NHEADS;
#pragma unroll
  for (int nt = 0; nt < 4; ++nt) {
#pragma unroll
    for (int jv = 0; jv < 8; ++jv) {
      const int qi = qt * 16 + jv + 8 * h;
      const int col = hh * HD + nt * 16 + (lane & 15);
      ao[(size_t)(bb * SEQL + qi) * DIMX + col] = (_Float16)(o[nt][jv] / l[jv]);
    }
  }
}

// ==================== Kernel 5: output projection (fp32 out) =================
__global__ void out_gemm(const _Float16* __restrict__ ao,
                         const _Float16* __restrict__ woh,
                         float* __restrict__ out) {
  const int wave = blockIdx.x * (blockDim.x >> 5) + (threadIdx.x >> 5);
  const int lane = threadIdx.x & 31;
  const int rt = wave >> 4;            // 128 row tiles of 32
  const int cg = wave & 15;            // 16 col groups of 64

  v8f c[2][4] = {};
  for (int kb = 0; kb < DIMX; kb += 32) {
    __builtin_prefetch(woh + (size_t)(cg * HD + (lane & 15)) * DIMX + kb + 128, 0, 0);
    __builtin_prefetch(ao + (size_t)(rt * 32 + (lane & 31)) * DIMX + kb + 128, 0, 0);
    Frag a0 = load_a(ao, DIMX, rt * 32, kb, lane);
    Frag a1 = load_a(ao, DIMX, rt * 32 + 16, kb, lane);
#pragma unroll
    for (int nt = 0; nt < 4; ++nt) {
      Frag b = load_bt(woh, DIMX, cg * HD + nt * 16, kb, lane);
      c[0][nt] = wmma_f16(a0, b, c[0][nt]);
      c[1][nt] = wmma_f16(a1, b, c[1][nt]);
    }
  }
  const int h = lane >> 4, col = lane & 15;
#pragma unroll
  for (int st = 0; st < 2; ++st)
#pragma unroll
    for (int nt = 0; nt < 4; ++nt)
#pragma unroll
      for (int jv = 0; jv < 8; ++jv)
        out[(size_t)(rt * 32 + st * 16 + jv + 8 * h) * DIMX + cg * HD + nt * 16 + col] =
            c[st][nt][jv];
}

// =============================================================================
extern "C" void kernel_launch(void* const* d_in, const int* in_sizes, int n_in,
                              void* d_out, int out_size, void* d_ws, size_t ws_size,
                              hipStream_t stream) {
  (void)in_sizes; (void)n_in; (void)out_size; (void)ws_size;
  const float* tokens = (const float*)d_in[0];
  const float* nw     = (const float*)d_in[1];
  const float* wq     = (const float*)d_in[2];
  const float* wk     = (const float*)d_in[3];
  const float* wv     = (const float*)d_in[4];
  const float* wo     = (const float*)d_in[5];
  float* out = (float*)d_out;

  // Workspace layout (~48 MB total)
  char* ws = (char*)d_ws;
  size_t off = 0;
  _Float16* w_h = (_Float16*)(ws + off); off += (size_t)4 * DIMX * DIMX * 2;   // 8 MB
  _Float16* xh  = (_Float16*)(ws + off); off += (size_t)NROWS * DIMX * 2;      // 8 MB
  _Float16* qh  = (_Float16*)(ws + off); off += (size_t)BATCH * NHEADS * SEQL * HD * 2;
  _Float16* kh  = (_Float16*)(ws + off); off += (size_t)BATCH * NHEADS * SEQL * HD * 2;
  _Float16* vh  = (_Float16*)(ws + off); off += (size_t)BATCH * NHEADS * SEQL * HD * 2;
  _Float16* ao  = (_Float16*)(ws + off); off += (size_t)NROWS * DIMX * 2;      // 8 MB

  // 0) weights -> f16
  cvt_weights<<<dim3(4 * DIMX * DIMX / 256), dim3(256), 0, stream>>>(wq, wk, wv, wo, w_h);
  // 1) RMSNorm -> f16 activations
  rmsnorm<<<dim3(NROWS), dim3(256), 0, stream>>>(tokens, nw, xh);
  // 2) QKV GEMMs: 3 mats * 128 row-tiles * 16 head-groups = 6144 waves, 4/block
  qkv_gemm<<<dim3(3 * (NROWS / 32) * (DIMX / HD) / 4), dim3(128), 0, stream>>>(
      xh, w_h, qh, kh, vh);
  // 3) RoPE on q (y=0) and k (y=1)
  rope_kernel<<<dim3(BATCH * NHEADS * SEQL * 32 / 256, 2), dim3(256), 0, stream>>>(qh, kh);
  // 4) causal flash attention: one wave per (bh, 16-query tile)
  attn_kernel<<<dim3(SEQL / 16, BATCH * NHEADS), dim3(32), 0, stream>>>(qh, kh, vh, ao);
  // 5) output projection -> fp32
  out_gemm<<<dim3((NROWS / 32) * (DIMX / HD) / 4), dim3(128), 0, stream>>>(
      ao, w_h + (size_t)3 * DIMX * DIMX, out);
}
